// GlobalBiDAFAttention_81071802679966
// MI455X (gfx1250) — matrix operand order repaired and checked by reading, not compiled, not run on hardware
//
#include <hip/hip_runtime.h>

typedef __attribute__((ext_vector_type(16))) __bf16 v16bf;
typedef __attribute__((ext_vector_type(8)))  float  v8f;

#define LCDIM 2048
#define LQDIM 128
#define HDIM  200
#define HP    224      // 7 k-chunks of 32
#define OUTW  1000
#define GIN   400
#define GOUT  200
#define GOUTP 208      // 13 n-blocks of 16

union FragBF { v16bf v; unsigned short u[16]; };

__device__ __forceinline__ unsigned short f2bf(float x) {
  unsigned u = __builtin_bit_cast(unsigned, x);
  unsigned r = u + 0x7FFFu + ((u >> 16) & 1u);   // round-to-nearest-even
  return (unsigned short)(r >> 16);
}
__device__ __forceinline__ unsigned f2bf2(float a, float b) {
  return (unsigned)f2bf(a) | ((unsigned)f2bf(b) << 16);
}

// A fragment, 16x32 bf16, source row-major [m][k], row stride ld.
// Two contiguous 16-byte runs per lane -> 2x ds_load_b128.
__device__ __forceinline__ v16bf loadA(const unsigned short* p, int ld, int lane) {
  FragBF f;
  const int m  = lane & 15;
  const int hb = (lane >> 4) ? 8 : 0;
  const unsigned short* row = p + m * ld;
#pragma unroll
  for (int v = 0; v < 8; ++v) {
    const int kl = (v < 4) ? (2 * v + hb) : (16 + 2 * (v - 4) + hb);
    f.u[2 * v]     = row[kl];
    f.u[2 * v + 1] = row[kl + 1];
  }
  return f.v;
}

// B fragment, 32x16 bf16, source stored N-major: element (k,n) = p[n*ld + k].
// Two contiguous 16-byte runs per lane -> 2x ds_load_b128.
__device__ __forceinline__ v16bf loadBt(const unsigned short* p, int ld, int lane) {
  FragBF f;
  const int n  = lane & 15;
  const int kb = (lane >> 4) * 8;
  const unsigned short* col = p + n * ld + kb;
#pragma unroll
  for (int v = 0; v < 8; ++v) {
    const int kl = (v < 4) ? (2 * v) : (16 + 2 * (v - 4));
    f.u[2 * v]     = col[kl];
    f.u[2 * v + 1] = col[kl + 1];
  }
  return f.v;
}

#define WMMA_BF16(A, B, C) \
  __builtin_amdgcn_wmma_f32_16x16x32_bf16(false, (A), false, (B), (short)0, (C), false, false)

static __device__ __forceinline__ v8f vzero8() { return (v8f){0, 0, 0, 0, 0, 0, 0, 0}; }

// ---------------------------------------------------------------------------
// Kernel 1: per batch -> column-softmax stats over Lc and t = s2m^T @ c,
// written TRANSPOSED ([h][q], bf16) to workspace. One WG / batch, 256 thr.
// ---------------------------------------------------------------------------
#define CT1 64

__global__ __launch_bounds__(256) void k1_col_t(
    const float* __restrict__ c, const float* __restrict__ q,
    const unsigned char* __restrict__ c_mask,
    const float* __restrict__ c_weight, const float* __restrict__ q_weight,
    const float* __restrict__ cq_weight, const float* __restrict__ bias,
    unsigned short* __restrict__ tT_out)
{
  __shared__ unsigned short qts[LQDIM * HP];    // [qj][h], q*cqw (B for S, N-major)
  __shared__ unsigned short cbf[CT1 * HP];      // [r][h]  (A for S)
  __shared__ unsigned short cbfT[HP * CT1];     // [h][r]  (B for t, N-major)
  __shared__ unsigned short etl[LQDIM * CT1];   // [q][r]  (A for t)
  __shared__ float pred[4 * LQDIM];             // per-mb partial col max / sum
  __shared__ float s1buf[LQDIM], colmax[LQDIM], colsum[LQDIM];
  __shared__ float s0buf[CT1], cml[CT1];

  const int b    = blockIdx.x;
  const int tid  = threadIdx.x;
  const int lane = tid & 31;
  const int wv   = tid >> 5;
  const int n    = lane & 15;
  const int half = lane >> 4;

  const float* qb = q + (size_t)b * LQDIM * HDIM;
  const float* cb = c + (size_t)b * LCDIM * HDIM;
  const unsigned char* cmb = c_mask + (size_t)b * LCDIM;
  const float bias0 = bias[0];

  for (int i = tid; i < LQDIM * HP / 2; i += 256) {
    const int j = (2 * i) / HP, h0 = (2 * i) % HP;
    const float a  = (h0 < HDIM)     ? qb[j * HDIM + h0] * cq_weight[h0]         : 0.f;
    const float b2 = (h0 + 1 < HDIM) ? qb[j * HDIM + h0 + 1] * cq_weight[h0 + 1] : 0.f;
    ((unsigned*)qts)[i] = f2bf2(a, b2);
  }
  if (tid < LQDIM) {
    float s = 0.f;
    for (int hh = 0; hh < HDIM; ++hh) s += qb[tid * HDIM + hh] * q_weight[hh];
    s1buf[tid]  = s;
    colmax[tid] = -3.0e38f;
    colsum[tid] = 0.f;
  }
  __syncthreads();

  v8f tacc[14];
#pragma unroll
  for (int i = 0; i < 14; ++i) tacc[i] = vzero8();

  const int mb  = wv & 3;
  const int nb0 = (wv >> 2) * 4;

  for (int pass = 0; pass < 2; ++pass) {
    for (int tile = 0; tile < LCDIM / CT1; ++tile) {
      const int r0 = tile * CT1;
      __syncthreads();                     // staging-buffer reuse guard
      for (int i = tid; i < CT1 * HP; i += 256) {
        const int r = i / HP, hh = i % HP;
        const float v = (hh < HDIM) ? cb[(size_t)(r0 + r) * HDIM + hh] : 0.f;
        const unsigned short bv = f2bf(v);
        cbf[i] = bv;
        cbfT[hh * CT1 + r] = bv;
      }
      if (tid < CT1) {
        float s = 0.f;
        for (int hh = 0; hh < HDIM; ++hh)
          s += cb[(size_t)(r0 + tid) * HDIM + hh] * c_weight[hh];
        s0buf[tid] = s;
        cml[tid]   = cmb[r0 + tid] ? 1.f : 0.f;
      }
      __syncthreads();

      // ---- S tile (64x128): wave -> rows 16*mb.., 4 n-blocks nb0..nb0+3
      v8f sa[4];
#pragma unroll
      for (int jj = 0; jj < 4; ++jj) sa[jj] = vzero8();
#pragma unroll
      for (int kc = 0; kc < 7; ++kc) {
        v16bf A = loadA(cbf + mb * 16 * HP + kc * 32, HP, lane);
#pragma unroll
        for (int jj = 0; jj < 4; ++jj) {
          v16bf Bf = loadBt(qts + (nb0 + jj) * 16 * HP + kc * 32, HP, lane);
          sa[jj] = WMMA_BF16(A, Bf, sa[jj]);
        }
      }

      // ---- column stats directly from D fragments
#pragma unroll
      for (int jj = 0; jj < 4; ++jj) {
        const int N = (nb0 + jj) * 16 + n;
        const float s1v = s1buf[N];
        float sv[8];
#pragma unroll
        for (int v = 0; v < 8; ++v) {
          const int M = mb * 16 + v + 8 * half;
          const float s = sa[jj][v] + s0buf[M] + s1v + bias0;
          sv[v] = (cml[M] != 0.f) ? s : -1e30f;
        }
        if (pass == 0) {
          float m = sv[0];
#pragma unroll
          for (int v = 1; v < 8; ++v) m = fmaxf(m, sv[v]);
          m = fmaxf(m, __shfl_xor(m, 16, 32));      // merge M-halves (same N)
          if (half == 0) pred[mb * LQDIM + N] = m;
        } else {
          const float cm2 = colmax[N];
          float ssum = 0.f;
          unsigned w[4];
#pragma unroll
          for (int p2 = 0; p2 < 4; ++p2) {
            const float e0 = __expf(sv[2 * p2] - cm2);       // masked rows -> 0
            const float e1 = __expf(sv[2 * p2 + 1] - cm2);
            ssum += e0 + e1;
            w[p2] = f2bf2(e0, e1);
          }
          // E^T row N, 8 contiguous r-values -> one ds_store_b128
          uint4 wq; wq.x = w[0]; wq.y = w[1]; wq.z = w[2]; wq.w = w[3];
          *reinterpret_cast<uint4*>(&etl[N * CT1 + mb * 16 + 8 * half]) = wq;
          ssum += __shfl_xor(ssum, 16, 32);
          if (half == 0) pred[mb * LQDIM + N] = ssum;
        }
      }
      __syncthreads();
      if (tid < LQDIM) {
        if (pass == 0) {
          float m = fmaxf(fmaxf(pred[tid], pred[LQDIM + tid]),
                          fmaxf(pred[2 * LQDIM + tid], pred[3 * LQDIM + tid]));
          colmax[tid] = fmaxf(colmax[tid], m);
        } else {
          colsum[tid] += pred[tid] + pred[LQDIM + tid] +
                         pred[2 * LQDIM + tid] + pred[3 * LQDIM + tid];
        }
      }
      if (pass == 1) {
        // t += E^T @ c_tile : wave wv owns q rows 16*wv..16*wv+15
#pragma unroll
        for (int kc = 0; kc < CT1 / 32; ++kc) {
          v16bf A = loadA(etl + wv * 16 * CT1 + kc * 32, CT1, lane);
          for (int nb = 0; nb < 14; ++nb) {
            v16bf Bf = loadBt(cbfT + nb * 16 * CT1 + kc * 32, CT1, lane);
            tacc[nb] = WMMA_BF16(A, Bf, tacc[nb]);
          }
        }
      }
    }
    __syncthreads();
  }

  // finalize: tT[h][q] = acc/colsum, packed uint4 global stores
  float cs[8];
#pragma unroll
  for (int v = 0; v < 8; ++v) cs[v] = colsum[wv * 16 + 8 * half + v];
  unsigned short* tb = tT_out + (size_t)b * HP * LQDIM;
  for (int nb = 0; nb < 14; ++nb) {
    const int hcol = nb * 16 + n;
    unsigned w[4];
#pragma unroll
    for (int p2 = 0; p2 < 4; ++p2)
      w[p2] = f2bf2(tacc[nb][2 * p2] / cs[2 * p2], tacc[nb][2 * p2 + 1] / cs[2 * p2 + 1]);
    uint4 wq; wq.x = w[0]; wq.y = w[1]; wq.z = w[2]; wq.w = w[3];
    *reinterpret_cast<uint4*>(&tb[hcol * LQDIM + wv * 16 + 8 * half]) = wq;
  }
}

// ---------------------------------------------------------------------------
// Kernel 2: per (batch, 128-row c-tile): S tile, in-register row softmax,
//           a = P1@q, b = P1@t, fused concat output.
// ---------------------------------------------------------------------------
#define CT2 128

__global__ __launch_bounds__(256) void k2_row_out(
    const float* __restrict__ c, const float* __restrict__ q,
    const unsigned char* __restrict__ q_mask,
    const float* __restrict__ c_weight, const float* __restrict__ q_weight,
    const float* __restrict__ cq_weight, const float* __restrict__ bias,
    const unsigned short* __restrict__ tT_in, float* __restrict__ out)
{
  __shared__ unsigned short qbuf[LQDIM * HP];   // [qj][h] unscaled q; later tT [h][q]
  __shared__ unsigned short rbuf[CT2 * HP];     // scaled c tile; later qT [h][q]
  __shared__ unsigned short p1l[CT2 * LQDIM];   // P1 bf16 (A for a/b)
  __shared__ float s0buf[CT2], s1buf[LQDIM];

  const int b    = blockIdx.x >> 4;
  const int tile = blockIdx.x & 15;
  const int r0   = tile * CT2;
  const int tid  = threadIdx.x;
  const int lane = tid & 31;
  const int wv   = tid >> 5;
  const int n    = lane & 15;
  const int half = lane >> 4;

  const float* qb = q + (size_t)b * LQDIM * HDIM;
  const float* cb = c + (size_t)b * LCDIM * HDIM;
  const float bias0 = bias[0];

  for (int i = tid; i < LQDIM * HP / 2; i += 256) {
    const int j = (2 * i) / HP, h0 = (2 * i) % HP;
    const float a  = (h0 < HDIM)     ? qb[j * HDIM + h0]     : 0.f;
    const float b2 = (h0 + 1 < HDIM) ? qb[j * HDIM + h0 + 1] : 0.f;
    ((unsigned*)qbuf)[i] = f2bf2(a, b2);
  }
  for (int i = tid; i < CT2 * HP / 2; i += 256) {
    const int r = (2 * i) / HP, h0 = (2 * i) % HP;
    const float a  = (h0 < HDIM)     ? cb[(size_t)(r0 + r) * HDIM + h0] * cq_weight[h0]         : 0.f;
    const float b2 = (h0 + 1 < HDIM) ? cb[(size_t)(r0 + r) * HDIM + h0 + 1] * cq_weight[h0 + 1] : 0.f;
    ((unsigned*)rbuf)[i] = f2bf2(a, b2);
  }
  if (tid < LQDIM) {
    float s = 0.f;
    for (int hh = 0; hh < HDIM; ++hh) s += qb[tid * HDIM + hh] * q_weight[hh];
    s1buf[tid] = s;
  }
  if (tid < CT2) {
    float s = 0.f;
    for (int hh = 0; hh < HDIM; ++hh) s += cb[(size_t)(r0 + tid) * HDIM + hh] * c_weight[hh];
    s0buf[tid] = s;
  }
  __syncthreads();

  // ---- S fragments: wave wv owns c rows 16wv.., all 8 q-blocks
  v8f sacc[8];
  float qm[8];
#pragma unroll
  for (int nb = 0; nb < 8; ++nb) {
    sacc[nb] = vzero8();
    qm[nb] = q_mask[(size_t)b * LQDIM + nb * 16 + n] ? 1.f : 0.f;
  }
#pragma unroll
  for (int kc = 0; kc < 7; ++kc) {
    v16bf A = loadA(rbuf + wv * 16 * HP + kc * 32, HP, lane);
#pragma unroll
    for (int nb = 0; nb < 8; ++nb) {
      v16bf Bf = loadBt(qbuf + nb * 16 * HP + kc * 32, HP, lane);
      sacc[nb] = WMMA_BF16(A, Bf, sacc[nb]);
    }
  }

  // ---- row softmax in registers (rows live in 16-lane halves)
#pragma unroll
  for (int v = 0; v < 8; ++v) {
    const int Ml = wv * 16 + v + 8 * half;
    const float base = s0buf[Ml] + bias0;
    float sv[8];
    float m = -3.0e38f;
#pragma unroll
    for (int nb = 0; nb < 8; ++nb) {
      const float s = sacc[nb][v] + base + s1buf[nb * 16 + n];
      sv[nb] = qm[nb] ? s : -1e30f;
      m = fmaxf(m, sv[nb]);
    }
    for (int off = 1; off < 16; off <<= 1) m = fmaxf(m, __shfl_xor(m, off, 32));
    float sum = 0.f;
    float ev[8];
#pragma unroll
    for (int nb = 0; nb < 8; ++nb) { ev[nb] = qm[nb] ? __expf(sv[nb] - m) : 0.f; sum += ev[nb]; }
    for (int off = 1; off < 16; off <<= 1) sum += __shfl_xor(sum, off, 32);
    const float inv = 1.f / sum;
#pragma unroll
    for (int nb = 0; nb < 8; ++nb)
      p1l[Ml * LQDIM + nb * 16 + n] = f2bf(ev[nb] * inv);
  }
  __syncthreads();

  // ---- build qT [h][q] in rbuf (scaled-c is dead), reading live qbuf
  for (int i = tid; i < HP * LQDIM; i += 256) {
    const int hh = i / LQDIM, j = i % LQDIM;
    rbuf[i] = qbuf[j * HP + hh];
  }
  __syncthreads();
  // ---- copy tT [h][q] into qbuf region (uint4 copies)
  {
    const uint4* src = reinterpret_cast<const uint4*>(tT_in + (size_t)b * HP * LQDIM);
    uint4* dst = reinterpret_cast<uint4*>(qbuf);
    for (int i = tid; i < HP * LQDIM / 8; i += 256) dst[i] = src[i];
  }
  __syncthreads();

  const unsigned short* qT = rbuf;   // [h][q]
  const unsigned short* tT = qbuf;   // [h][q]

  // ---- a = P1 @ q, b = P1 @ t; fused output per h-block
  for (int nb = 0; nb < 14; ++nb) {
    v8f accA = vzero8();
    v8f accB = vzero8();
#pragma unroll
    for (int kc = 0; kc < 4; ++kc) {
      v16bf A  = loadA(p1l + wv * 16 * LQDIM + kc * 32, LQDIM, lane);
      v16bf Ba = loadBt(qT + nb * 16 * LQDIM + kc * 32, LQDIM, lane);
      v16bf Bb = loadBt(tT + nb * 16 * LQDIM + kc * 32, LQDIM, lane);
      accA = WMMA_BF16(A, Ba, accA);
      accB = WMMA_BF16(A, Bb, accB);
    }
    const int hcol = nb * 16 + n;
    if (hcol < HDIM) {
#pragma unroll
      for (int v = 0; v < 8; ++v) {
        const int row = r0 + wv * 16 + v + 8 * half;
        const float cv = cb[(size_t)row * HDIM + hcol];
        const size_t ob = ((size_t)b * LCDIM + row) * OUTW;
        out[ob + hcol]       = cv;
        out[ob + 200 + hcol] = accA[v];
        out[ob + 400 + hcol] = cv * accA[v];
        out[ob + 600 + hcol] = cv * accB[v];
      }
    }
  }
}

// ---------------------------------------------------------------------------
// Kernel 3: global_sim = (q_global ⊙ c_conv) @ proj_w + proj_b
// ---------------------------------------------------------------------------
__global__ __launch_bounds__(256) void k3_gsim(
    const float* __restrict__ q_global, const float* __restrict__ c_conv,
    const float* __restrict__ proj_w, const float* __restrict__ proj_b,
    float* __restrict__ out)
{
  __shared__ unsigned short ael[128 * 32];     // [r][k] (A)
  __shared__ unsigned short welT[GOUTP * 32];  // [n][k] (B, N-major)

  const int b    = blockIdx.x >> 4;
  const int tile = blockIdx.x & 15;
  const int r0   = tile * 128;
  const int tid  = threadIdx.x;
  const int lane = tid & 31;
  const int wv   = tid >> 5;
  const int n    = lane & 15;
  const int half = lane >> 4;

  v8f acc[13];
#pragma unroll
  for (int i = 0; i < 13; ++i) acc[i] = vzero8();

  for (int kc = 0; kc < 13; ++kc) {
    __syncthreads();
    if (kc + 1 < 13 && tid < 128)   // hint next K-chunk toward the caches
      __builtin_prefetch(&c_conv[((size_t)b * LCDIM + r0 + tid) * GIN + (kc + 1) * 32], 0, 1);
    for (int i = tid; i < 128 * 32 / 2; i += 256) {
      const int r = (2 * i) >> 5, k0 = ((2 * i) & 31) + kc * 32;
      const size_t rb = ((size_t)b * LCDIM + r0 + r) * GIN;
      const float a  = (k0 < GIN)     ? c_conv[rb + k0] * q_global[(size_t)b * GIN + k0]         : 0.f;
      const float b2 = (k0 + 1 < GIN) ? c_conv[rb + k0 + 1] * q_global[(size_t)b * GIN + k0 + 1] : 0.f;
      ((unsigned*)ael)[i] = f2bf2(a, b2);
    }
    for (int i = tid; i < GOUTP * 32; i += 256) {
      const int nn = i >> 5, kl = i & 31, k = kl + kc * 32;
      welT[i] = f2bf((k < GIN && nn < GOUT) ? proj_w[(size_t)k * GOUT + nn] : 0.f);
    }
    __syncthreads();
    v16bf A = loadA(ael + wv * 16 * 32, 32, lane);
#pragma unroll
    for (int nb = 0; nb < 13; ++nb) {
      v16bf Bf = loadBt(welT + nb * 16 * 32, 32, lane);
      acc[nb] = WMMA_BF16(A, Bf, acc[nb]);
    }
  }

#pragma unroll
  for (int nb = 0; nb < 13; ++nb) {
    const int hcol = nb * 16 + n;
    if (hcol < GOUT) {
#pragma unroll
      for (int v = 0; v < 8; ++v) {
        const int row = r0 + wv * 16 + v + 8 * half;
        out[((size_t)b * LCDIM + row) * OUTW + 800 + hcol] = acc[nb][v] + proj_b[hcol];
      }
    }
  }
}

// ---------------------------------------------------------------------------
extern "C" void kernel_launch(void* const* d_in, const int* in_sizes, int n_in,
                              void* d_out, int out_size, void* d_ws, size_t ws_size,
                              hipStream_t stream) {
  (void)in_sizes; (void)n_in; (void)out_size; (void)ws_size;
  const float*         c        = (const float*)d_in[0];
  const float*         q        = (const float*)d_in[1];
  const unsigned char* c_mask   = (const unsigned char*)d_in[2];
  const unsigned char* q_mask   = (const unsigned char*)d_in[3];
  const float*         q_global = (const float*)d_in[4];
  const float*         c_conv   = (const float*)d_in[5];
  const float*         c_weight = (const float*)d_in[6];
  const float*         q_weight = (const float*)d_in[7];
  const float*         cq_w     = (const float*)d_in[8];
  const float*         bias     = (const float*)d_in[9];
  const float*         proj_w   = (const float*)d_in[10];
  const float*         proj_b   = (const float*)d_in[11];
  float*               out      = (float*)d_out;
  unsigned short*      t_ws     = (unsigned short*)d_ws;  // 64 * 224 * 128 bf16 = 3.5 MB

  k1_col_t  <<<64,   256, 0, stream>>>(c, q, c_mask, c_weight, q_weight, cq_w, bias, t_ws);
  k2_row_out<<<1024, 256, 0, stream>>>(c, q, q_mask, c_weight, q_weight, cq_w, bias, t_ws, out);
  k3_gsim   <<<1024, 256, 0, stream>>>(q_global, c_conv, proj_w, proj_b, out);
}
